// Attention_64974265254303
// MI455X (gfx1250) — matrix operand order repaired and checked by reading, not compiled, not run on hardware
//
#include <hip/hip_runtime.h>

// ---------------------------------------------------------------------------
// MI455X (gfx1250) attention pipeline, round 3.
// - All GEMM operands pre-converted to bf16 (halves HBM traffic vs f32).
// - All B operands pre-arranged [n][k] so LDS staging is a raw byte copy.
// - GEMM: 128x128x64 tiles, double-buffered LDS, GLOBAL_LOAD_ASYNC_TO_LDS_B128
//   (ASYNCcnt pipelined) when available, v_wmma_f32_16x16x32_bf16 compute.
// ---------------------------------------------------------------------------

typedef __attribute__((ext_vector_type(16))) __bf16 v16bf;
typedef __attribute__((ext_vector_type(8)))  float  v8f;

// Pointer types matching the async-to-LDS builtin prototype "vV4i*1V4i*3IiIi"
typedef int v4i_chunk __attribute__((vector_size(16)));
typedef __attribute__((address_space(1))) v4i_chunk* g128_t;
typedef __attribute__((address_space(3))) v4i_chunk* l128_t;

#define BM 128
#define BN 128
#define BK 64
#define LDSP 72                   // bf16 pitch: 144B rows, 16B aligned, conflict-free
#define TS   (128 * LDSP)         // shorts per tile

#if defined(__has_builtin)
#  if __has_builtin(__builtin_amdgcn_global_load_async_to_lds_b128)
#    define USE_ASYNC_LDS 1
#  endif
#endif
#ifndef USE_ASYNC_LDS
#  define USE_ASYNC_LDS 0
#endif

union FragAB { unsigned int u[8]; v16bf v; };

__device__ __forceinline__ unsigned short f32_to_bf16(float f) {
    unsigned int u = __float_as_uint(f);
    u += 0x7FFFu + ((u >> 16) & 1u);   // round-to-nearest-even
    return (unsigned short)(u >> 16);
}
__device__ __forceinline__ unsigned int pack_bf16(float lo, float hi) {
    return (unsigned int)f32_to_bf16(lo) | ((unsigned int)f32_to_bf16(hi) << 16);
}

template <int N>
__device__ __forceinline__ void wait_async() {
#if USE_ASYNC_LDS
#  if __has_builtin(__builtin_amdgcn_s_wait_asynccnt)
    __builtin_amdgcn_s_wait_asynccnt(N);
#  else
    asm volatile("s_wait_asynccnt %0" :: "n"(N) : "memory");
#  endif
#endif
}

// Copy one 128-row x 64-col bf16 tile (row stride ld, in elements) into LDS.
// 1024 chunks of 16B; 256 threads x 4 chunks. Async (ASYNCcnt) when available.
__device__ __forceinline__ void stage_tile(const unsigned short* __restrict__ g,
                                           long long ld, unsigned short* l, int tid) {
#pragma unroll
    for (int p = 0; p < 4; ++p) {
        int i   = p * 256 + tid;
        int row = i >> 3;
        int c   = i & 7;
        const unsigned short* gs = g + (long long)row * ld + c * 8;
        unsigned short*       ls = l + row * LDSP + c * 8;
#if USE_ASYNC_LDS
        // AS1 from flat (identical bits); AS3 = low 32 bits of flat LDS address
        __builtin_amdgcn_global_load_async_to_lds_b128(
            (g128_t)(unsigned long long)gs,
            (l128_t)(unsigned int)(unsigned long long)ls,
            0, 0);
#else
        *(uint4*)ls = *(const uint4*)gs;
#endif
    }
}

struct GemmParams {
    const unsigned short* A;    // M x K bf16, row stride lda
    const unsigned short* A2;   // optional K-concat second source
    long long lda, strideAb, kSplit;
    const unsigned short* B;    // [n][k] bf16, row stride ldb
    long long ldb, strideBb;
    float*          Cf;         // epi 0/1 output
    unsigned short* Cbf;        // epi 2 output
    long long ldc, strideCb;
    int M, N, K;
    int epi;                    // 0: f32   1: tanh(x + bias[n]) f32   2: bf16
    const float* bias;
};

__global__ __launch_bounds__(256)
void gemm_bf16_wmma_kernel(GemmParams p) {
    extern __shared__ __align__(16) unsigned short smem[];   // [A0][B0][A1][B1]

    const int tid  = threadIdx.x;
    const int lane = tid & 31;
    const int wave = tid >> 5;
    const int wm   = wave & 3;   // 4 waves x 32 rows
    const int wn   = wave >> 2;  // 2 waves x 64 cols
    const int half = lane >> 4;
    const int l15  = lane & 15;

    const long long batch = blockIdx.z;
    const unsigned short* A  = p.A + batch * p.strideAb;
    const unsigned short* A2 = p.A2 ? (p.A2 + batch * p.strideAb) : nullptr;
    const unsigned short* B  = p.B + batch * p.strideBb;

    const long long rowBase = (long long)blockIdx.y * BM;
    const long long colBase = (long long)blockIdx.x * BN;

    auto aTile = [&](long long kk) -> const unsigned short* {
        if (A2 && kk >= p.kSplit) return A2 + rowBase * p.lda + (kk - p.kSplit);
        return A + rowBase * p.lda + kk;
    };

    v8f acc[2][4];
#pragma unroll
    for (int mt = 0; mt < 2; ++mt)
#pragma unroll
        for (int nt = 0; nt < 4; ++nt)
#pragma unroll
            for (int j = 0; j < 8; ++j) acc[mt][nt][j] = 0.0f;

    const int nkt = p.K / BK;

    // Prologue: stage tile 0 into buffer 0 (8 async instructions per wave).
    stage_tile(aTile(0), p.lda, smem, tid);
    stage_tile(B + colBase * p.ldb, p.ldb, smem + TS, tid);

    for (int kt = 0; kt < nkt; ++kt) {
        const int cur = kt & 1;
        if (kt + 1 < nkt) {
            const long long kk = (long long)(kt + 1) * BK;
            unsigned short* nb = smem + (cur ^ 1) * 2 * TS;
            stage_tile(aTile(kk), p.lda, nb, tid);
            stage_tile(B + colBase * p.ldb + kk, p.ldb, nb + TS, tid);
            wait_async<8>();   // tile kt done; tile kt+1 (8 asyncs) stays in flight
        } else {
            wait_async<0>();
        }
        __syncthreads();

        const unsigned short* lsA = smem + cur * 2 * TS;
        const unsigned short* lsB = lsA + TS;

#pragma unroll
        for (int k0 = 0; k0 < BK; k0 += 32) {
            FragAB bf[4];
#pragma unroll
            for (int nt = 0; nt < 4; ++nt) {
                int n = wn * 64 + nt * 16 + l15;
                const unsigned int* s =
                    (const unsigned int*)&lsB[n * LDSP + k0 + 16 * half];
#pragma unroll
                for (int j = 0; j < 8; ++j) bf[nt].u[j] = s[j];
            }
            FragAB af[2];
#pragma unroll
            for (int mt = 0; mt < 2; ++mt) {
                int m = wm * 32 + mt * 16 + l15;
                const unsigned int* s = (const unsigned int*)&lsA[m * LDSP + k0];
#pragma unroll
                for (int j = 0; j < 4; ++j) {
                    af[mt].u[j]     = s[8 * half + j];
                    af[mt].u[j + 4] = s[8 * half + 4 + j];
                }
            }
#pragma unroll
            for (int mt = 0; mt < 2; ++mt)
#pragma unroll
                for (int nt = 0; nt < 4; ++nt)
                    acc[mt][nt] = __builtin_amdgcn_wmma_f32_16x16x32_bf16(
                        false, af[mt].v, false, bf[nt].v,
                        (short)0, acc[mt][nt], false, false);
        }
        __syncthreads();   // everyone done reading buf[cur] before it is re-staged
    }

    // Epilogue (C/D layout: vgpr j -> row j + 8*lanegroup, col lane&15)
#pragma unroll
    for (int mt = 0; mt < 2; ++mt)
#pragma unroll
        for (int nt = 0; nt < 4; ++nt) {
            long long col = colBase + wn * 64 + nt * 16 + l15;
            float bv = (p.epi == 1) ? p.bias[col] : 0.0f;
#pragma unroll
            for (int j = 0; j < 8; ++j) {
                long long row = rowBase + wm * 32 + mt * 16 + j + 8 * half;
                float v = acc[mt][nt][j];
                if (p.epi == 2) {
                    unsigned short* Cb = p.Cbf + batch * p.strideCb;
                    Cb[row * p.ldc + col] = f32_to_bf16(v);
                } else {
                    float* Cf = p.Cf + batch * p.strideCb;
                    if (p.epi == 1) v = tanhf(v + bv);
                    Cf[row * p.ldc + col] = v;
                }
            }
        }
}

// ---------------------------------------------------------------------------
// Pre-pass: f32 -> bf16 flat convert (float4 per thread).
// ---------------------------------------------------------------------------
__global__ __launch_bounds__(256)
void cvt_bf16_flat_kernel(const float* __restrict__ in,
                          unsigned short* __restrict__ out) {
    long long i = ((long long)blockIdx.x * 256 + threadIdx.x) * 4;
    float4 x = *(const float4*)(in + i);
    uint2 o;
    o.x = pack_bf16(x.x, x.y);
    o.y = pack_bf16(x.z, x.w);
    *(uint2*)(out + i) = o;
}

// ---------------------------------------------------------------------------
// Pre-pass: tiled f32 -> bf16 transpose.  out[c][r] = in[r][c]
// in: rows x cols (row stride ldin), out: cols x rows (row stride ldout).
// ---------------------------------------------------------------------------
__global__ __launch_bounds__(256)
void transpose_bf16_kernel(const float* __restrict__ in, long long ldin, long long strideInB,
                           unsigned short* __restrict__ out, long long ldout, long long strideOutB) {
    __shared__ float tile[32][33];
    const float*    I = in  + blockIdx.z * strideInB;
    unsigned short* O = out + blockIdx.z * strideOutB;
    const int r0 = blockIdx.y * 32, c0 = blockIdx.x * 32;
    const int tx = threadIdx.x & 31, ty = threadIdx.x >> 5;
#pragma unroll
    for (int p = 0; p < 4; ++p)
        tile[ty + p * 8][tx] = I[(long long)(r0 + ty + p * 8) * ldin + c0 + tx];
    __syncthreads();
#pragma unroll
    for (int p = 0; p < 4; ++p)
        O[(long long)(c0 + ty + p * 8) * ldout + (r0 + tx)] =
            f32_to_bf16(tile[tx][ty + p * 8]);
}

// ---------------------------------------------------------------------------
// Masked softmax over S (one wave per (t,b) row); writes bf16 `a`.
// ---------------------------------------------------------------------------
__global__ __launch_bounds__(256)
void softmax_mask_kernel(const float* __restrict__ score,
                         unsigned short* __restrict__ a,
                         const int* __restrict__ source, int Bn, int S) {
    int row  = blockIdx.x * 8 + (threadIdx.x >> 5);
    int lane = threadIdx.x & 31;
    int b    = row % Bn;
    const float*    r  = score + (long long)row * S;
    unsigned short* ro = a     + (long long)row * S;

    float v[32];
    float m = -3.402823466e38f;
#pragma unroll
    for (int i = 0; i < 32; ++i) {
        v[i] = r[i * 32 + lane];
        m = fmaxf(m, v[i]);
    }
#pragma unroll
    for (int off = 16; off; off >>= 1) m = fmaxf(m, __shfl_xor(m, off, 32));

    float sum = 0.0f;
#pragma unroll
    for (int i = 0; i < 32; ++i) {
        int s = i * 32 + lane;
        float e = (source[(long long)s * Bn + b] == 0) ? 0.0f : expf(v[i] - m);
        v[i] = e;
        sum += e;
    }
#pragma unroll
    for (int off = 16; off; off >>= 1) sum += __shfl_xor(sum, off, 32);

    float inv = 1.0f / sum;   // fully-masked row -> NaN, like the reference
#pragma unroll
    for (int i = 0; i < 32; ++i) ro[i * 32 + lane] = f32_to_bf16(v[i] * inv);
}

// ---------------------------------------------------------------------------
extern "C" void kernel_launch(void* const* d_in, const int* in_sizes, int n_in,
                              void* d_out, int out_size, void* d_ws, size_t ws_size,
                              hipStream_t stream) {
    (void)in_sizes; (void)n_in; (void)out_size; (void)ws_size;

    const float* ht     = (const float*)d_in[0];   // (T, B, H)
    const float* hs     = (const float*)d_in[1];   // (S, B, H)
    const int*   source = (const int*)  d_in[2];   // (S, B)
    const float* W_a    = (const float*)d_in[3];   // (H, H)
    const float* W_c    = (const float*)d_in[4];   // (2H, O)
    const float* bias   = (const float*)d_in[5];   // (O,)
    float*       out    = (float*)d_out;           // (T, B, O)

    const int T = 1024, S = 1024, Bn = 32, H = 1024, O = 1024;
    const size_t NTB = (size_t)T * Bn * H;         // 33,554,432

    // Workspace: f32 score first (alignment), then bf16 buffers.
    float* score = (float*)d_ws;                                   // T*B*S f32
    unsigned short* ht_bf   = (unsigned short*)(score + NTB);
    unsigned short* hs_bf   = ht_bf   + NTB;
    unsigned short* hsT_bf  = hs_bf   + NTB;                       // (B, H, S)
    unsigned short* keys_bf = hsT_bf  + NTB;                       // (S, B, H)
    unsigned short* a_bf    = keys_bf + NTB;                       // (T, B, S)
    unsigned short* c_bf    = a_bf    + NTB;                       // (T, B, H)
    unsigned short* WaT     = c_bf    + NTB;                       // (H, H)   [l][h]
    unsigned short* WcT     = WaT + (size_t)H * H;                 // (O, 2H)  [o][k]

    dim3 blk(256);
    const unsigned int gemmLds = 4u * TS * sizeof(unsigned short);  // 73,728 B

    // ---- Pre-pass: bf16 conversions / transposes --------------------------
    cvt_bf16_flat_kernel<<<dim3(NTB / 1024), blk, 0, stream>>>(ht, ht_bf);
    cvt_bf16_flat_kernel<<<dim3(NTB / 1024), blk, 0, stream>>>(hs, hs_bf);
    // hsT[b][h][s] = hs[s][b][h]
    transpose_bf16_kernel<<<dim3(H / 32, S / 32, Bn), blk, 0, stream>>>(
        hs, (long long)Bn * H, H, hsT_bf, S, (long long)H * S);
    // WaT[l][h] = W_a[h][l]
    transpose_bf16_kernel<<<dim3(H / 32, H / 32, 1), blk, 0, stream>>>(
        W_a, H, 0, WaT, H, 0);
    // WcT[o][k] = W_c[k][o]
    transpose_bf16_kernel<<<dim3(O / 32, (2 * H) / 32, 1), blk, 0, stream>>>(
        W_c, O, 0, WcT, 2 * H, 0);

    // ---- 1) keys = hs @ W_a  ->  bf16 (S*B x H) ---------------------------
    {
        GemmParams p{};
        p.A = hs_bf; p.lda = H; p.strideAb = 0; p.kSplit = 0;
        p.B = WaT;   p.ldb = H; p.strideBb = 0;
        p.Cbf = keys_bf; p.ldc = H; p.strideCb = 0;
        p.M = S * Bn; p.N = H; p.K = H; p.epi = 2;
        gemm_bf16_wmma_kernel<<<dim3(H / BN, (S * Bn) / BM, 1), blk, gemmLds, stream>>>(p);
    }
    // ---- 2) score[t,b,s] = ht . keys (batched over b) -> f32 --------------
    {
        GemmParams p{};
        p.A = ht_bf;   p.lda = (long long)Bn * H; p.strideAb = H; p.kSplit = 0;
        p.B = keys_bf; p.ldb = (long long)Bn * H; p.strideBb = H;   // [n=s][k=h]
        p.Cf = score;  p.ldc = (long long)Bn * S; p.strideCb = S;
        p.M = T; p.N = S; p.K = H; p.epi = 0;
        gemm_bf16_wmma_kernel<<<dim3(S / BN, T / BM, Bn), blk, gemmLds, stream>>>(p);
    }
    // ---- 3) masked softmax -> bf16 a --------------------------------------
    softmax_mask_kernel<<<dim3((T * Bn) / 8), blk, 0, stream>>>(score, a_bf, source, Bn, S);

    // ---- 4) c[t,b,h] = a . hs (batched) -> bf16 ---------------------------
    {
        GemmParams p{};
        p.A = a_bf;   p.lda = (long long)Bn * S; p.strideAb = S; p.kSplit = 0;
        p.B = hsT_bf; p.ldb = S; p.strideBb = (long long)H * S;     // [n=h][k=s]
        p.Cbf = c_bf; p.ldc = (long long)Bn * H; p.strideCb = H;
        p.M = T; p.N = H; p.K = S; p.epi = 2;
        gemm_bf16_wmma_kernel<<<dim3(H / BN, T / BM, Bn), blk, gemmLds, stream>>>(p);
    }
    // ---- 5) out = tanh(concat(c, ht) @ W_c + b) ---------------------------
    {
        GemmParams p{};
        p.A = c_bf; p.A2 = ht_bf; p.lda = H; p.strideAb = 0; p.kSplit = H;
        p.B = WcT;  p.ldb = 2 * H; p.strideBb = 0;                  // [n=o][k]
        p.Cf = out; p.ldc = O; p.strideCb = 0;
        p.M = T * Bn; p.N = O; p.K = 2 * H; p.epi = 1; p.bias = bias;
        gemm_bf16_wmma_kernel<<<dim3(O / BN, (T * Bn) / BM, 1), blk, gemmLds, stream>>>(p);
    }
}
